// SegmentTreeTransformer_74096775791001
// MI455X (gfx1250) — matrix operand order repaired and checked by reading, not compiled, not run on hardware
//
#include <hip/hip_runtime.h>

typedef _Float16 half_t;
typedef __attribute__((ext_vector_type(16))) _Float16 v16h;
typedef __attribute__((ext_vector_type(8)))  float    v8f;
typedef __attribute__((ext_vector_type(4)))  unsigned v4u;
typedef __attribute__((ext_vector_type(8)))  int      v8i;
typedef __attribute__((ext_vector_type(4)))  int      v4i;

constexpr int D     = 128;
constexpr int H     = 8;
constexpr int DK    = 16;
constexpr int DFF   = 512;
constexpr int NL    = 4;
constexpr int NCLS  = 10;
constexpr int NN    = 131072;
constexpr int NLEAF = 65536;
constexpr int EDGES = 524288 + 131072;   // ERAND + N self loops
constexpr int NREAD = 4096;

#if defined(__gfx1250__) && __has_builtin(__builtin_amdgcn_tensor_load_to_lds)
#define HAVE_TDM 1
#else
#define HAVE_TDM 0
#endif

// ---------------------------------------------------------------- fragment IO

union H16U { uint4 q[2]; half_t h[16]; };
union PairU { unsigned u; half_t h[2]; };

// A-matrix 16x32 f16 fragment per ISA layout:
// lanes 0-15: M=lane, VGPR j<4 -> K=2j,2j+1 ; j>=4 -> K=8+2j,9+2j
// lanes 16-31: same +8 on K
__device__ inline v16h load_a_f16(const half_t* __restrict__ A, int lda, int r0,
                                  int k0, int lane) {
  int hl = lane >> 4, m = lane & 15;
  const half_t* base = A + (size_t)(r0 + m) * lda + k0;
  v16h a;
#pragma unroll
  for (int j = 0; j < 8; ++j) {
    int kb = ((j < 4) ? (2 * j) : (8 + 2 * j)) + 8 * hl;
    PairU p;
    p.u = *reinterpret_cast<const unsigned*>(base + kb);
    a[2 * j]     = p.h[0];
    a[2 * j + 1] = p.h[1];
  }
  return a;
}

// Same fragment shape but from an LDS tile (base points into shared memory).
__device__ inline v16h load_a_lds(const half_t* tile, int ldt, int m0, int k0,
                                  int lane) {
  int hl = lane >> 4, m = lane & 15;
  const half_t* base = tile + (m0 + m) * ldt + k0;
  v16h a;
#pragma unroll
  for (int j = 0; j < 8; ++j) {
    int kb = ((j < 4) ? (2 * j) : (8 + 2 * j)) + 8 * hl;
    a[2 * j]     = base[kb];
    a[2 * j + 1] = base[kb + 1];
  }
  return a;
}

// B-matrix 32x16 fragment from pre-packed weights: each block is 512 halves,
// lane's 16 halves contiguous -> two b128 loads.
__device__ inline v16h load_b_packed(const half_t* __restrict__ Wp, int block,
                                     int lane) {
  const uint4* p =
      reinterpret_cast<const uint4*>(Wp + (size_t)block * 512 + lane * 16);
  H16U u;
  u.q[0] = p[0];
  u.q[1] = p[1];
  v16h b;
#pragma unroll
  for (int i = 0; i < 16; ++i) b[i] = u.h[i];
  return b;
}

__device__ inline v8f wmma_f16(v16h a, v16h b, v8f c) {
  return __builtin_amdgcn_wmma_f32_16x16x32_f16(false, a, false, b, (short)0, c,
                                                false, false);
}

// ------------------------------------------------- Tensor Data Mover staging
#if HAVE_TDM
// 2D TDM load: rows x 128 f16 tile, row stride `stride_elems`, into LDS byte
// offset `lds_off` (dynamic LDS starts at 0). D# per ISA 08_async_tensor §8.
__device__ inline void tdm_load_2d(unsigned lds_off, const half_t* gsrc,
                                   unsigned rows, unsigned stride_elems) {
  unsigned long long ga = (unsigned long long)(const void*)gsrc;
  v4u g0;
  g0[0] = 1u;                                           // count=1 valid user D#
  g0[1] = lds_off;                                      // lds_addr (bytes)
  g0[2] = (unsigned)ga;                                 // global_addr[31:0]
  g0[3] = (unsigned)((ga >> 32) & 0x01FFFFFFu) | 0x80000000u; // [56:32]+type=2
  v8i g1;
  g1[0] = 0x10000;                 // data_size=1 (2 bytes), no mask/pad/iter
  g1[1] = (int)(128u << 16);       // tensor_dim0 = 128  -> bits[79:48] lo
  g1[2] = (int)(rows << 16);       // tensor_dim0 hi=0 | tensor_dim1 lo = rows
  g1[3] = (int)(128u << 16);       // tensor_dim1 hi=0 | tile_dim0 = 128
  g1[4] = (int)rows;               // tile_dim1 = rows | tile_dim2 = 0
  g1[5] = (int)stride_elems;       // tensor_dim0_stride lo
  g1[6] = 0;                       // stride hi | tensor_dim1_stride lo
  g1[7] = 0;
  v4i z4 = {0, 0, 0, 0};
#if __clang_major__ >= 23
  v8i z8 = {0, 0, 0, 0, 0, 0, 0, 0};
  __builtin_amdgcn_tensor_load_to_lds(g0, g1, z4, z4, z8, 0);
#else
  __builtin_amdgcn_tensor_load_to_lds(g0, g1, z4, z4, 0);
#endif
}
#endif

// Stage `rows` x 128 f16 rows (contiguous, stride 128) into LDS at offset 0.
// Caller must __syncthreads() afterwards.
__device__ inline void stage_rows_to_lds(half_t* lds_dst, const half_t* gsrc,
                                         int rows) {
#if HAVE_TDM
  if (threadIdx.x < 32) {
    tdm_load_2d(0u, gsrc, (unsigned)rows, (unsigned)D);
    __builtin_amdgcn_s_wait_tensorcnt(0);
  }
#else
  int total = rows * D / 8;  // uint4 elements
  const uint4* s = reinterpret_cast<const uint4*>(gsrc);
  uint4* dp = reinterpret_cast<uint4*>(lds_dst);
  for (int i = threadIdx.x; i < total; i += 256) dp[i] = s[i];
#endif
}

// ------------------------------------------------------------- weight packing
// Pack row-major f32 W[K,O] into B-fragment blocks: block = kt*(O/16)+nt,
// within block: lane*16 + (2j+t), value = W[kt*32 + ((j<4?2j:8+2j)+8*hl) + t][nt*16 + n]
__global__ __launch_bounds__(256) void pack_w_kernel(const float* __restrict__ W,
                                                     half_t* __restrict__ Wp,
                                                     int K, int O) {
  int p = blockIdx.x * 256 + threadIdx.x;
  if (p >= K * O) return;
  int block  = p >> 9;
  int within = p & 511;
  int lane   = within >> 4;
  int idx    = within & 15;
  int j = idx >> 1, t = idx & 1;
  int nct = O >> 4;
  int kt = block / nct, nt = block % nct;
  int hl = lane >> 4, n = lane & 15;
  int kb = ((j < 4) ? (2 * j) : (8 + 2 * j)) + 8 * hl;
  Wp[p] = (half_t)W[(size_t)(kt * 32 + kb + t) * O + nt * 16 + n];
}

// ------------------------------------------------------------------ utilities
__global__ __launch_bounds__(256) void zero_f32_kernel(float* p, int n) {
  int i = blockIdx.x * 256 + threadIdx.x;
  if (i < n) p[i] = 0.0f;
}
__global__ __launch_bounds__(256) void zero_u32_kernel(unsigned* p, int n) {
  int i = blockIdx.x * 256 + threadIdx.x;
  if (i < n) p[i] = 0u;
}
__global__ __launch_bounds__(256) void zero_pair_kernel(float* h, half_t* h16,
                                                        int n) {
  int i = blockIdx.x * 256 + threadIdx.x;
  if (i < n) { h[i] = 0.0f; h16[i] = (half_t)0.0f; }
}

// ------------------------------------------------------------ embedding + LN
__global__ __launch_bounds__(256) void embed_kernel(
    const int* __restrict__ x, const int* __restrict__ pos,
    const int* __restrict__ leaf_ids, const float* __restrict__ et,
    const float* __restrict__ g, const float* __restrict__ b,
    float* __restrict__ h, half_t* __restrict__ h16) {
  int lane = threadIdx.x & 31, wave = threadIdx.x >> 5;
  int li = blockIdx.x * 8 + wave;
  if (li >= NLEAF) return;
  int   node = leaf_ids[li];
  int   tok  = x[li];
  float p    = (float)pos[li];
  float vals[4];
  int j0 = lane * 4;
#pragma unroll
  for (int t = 0; t < 4; ++t) {
    int   j    = j0 + t;
    float freq = __expf(-(float)(j & ~1) * (9.210340371976184f / 128.0f));
    float ang  = p * freq;
    float pe   = (j & 1) ? __cosf(ang) : __sinf(ang);
    vals[t]    = et[(size_t)tok * D + j] * 11.313708498984761f + pe;
  }
  float s = vals[0] + vals[1] + vals[2] + vals[3];
#pragma unroll
  for (int o = 16; o; o >>= 1) s += __shfl_xor(s, o);
  float mean = s * (1.0f / 128.0f);
  float vs = 0.0f;
#pragma unroll
  for (int t = 0; t < 4; ++t) { float d = vals[t] - mean; vs += d * d; }
#pragma unroll
  for (int o = 16; o; o >>= 1) vs += __shfl_xor(vs, o);
  float rstd = rsqrtf(vs * (1.0f / 128.0f) + 1e-5f);
#pragma unroll
  for (int t = 0; t < 4; ++t) {
    int   j  = j0 + t;
    float ov = (vals[t] - mean) * rstd * g[j] + b[j];
    h[(size_t)node * D + j]   = ov;
    h16[(size_t)node * D + j] = (half_t)ov;
  }
}

// -------------------------------------------------------- residual + LN + f16
__global__ __launch_bounds__(256) void resln_kernel(
    const float* __restrict__ hin, const float* __restrict__ add,
    const float* __restrict__ g, const float* __restrict__ b,
    float* __restrict__ hout, half_t* __restrict__ h16) {
  int lane = threadIdx.x & 31, wave = threadIdx.x >> 5;
  int row = blockIdx.x * 8 + wave;
  size_t base = (size_t)row * D;
  int j0 = lane * 4;
  float vals[4];
#pragma unroll
  for (int t = 0; t < 4; ++t) vals[t] = hin[base + j0 + t] + add[base + j0 + t];
  float s = vals[0] + vals[1] + vals[2] + vals[3];
#pragma unroll
  for (int o = 16; o; o >>= 1) s += __shfl_xor(s, o);
  float mean = s * (1.0f / 128.0f);
  float vs = 0.0f;
#pragma unroll
  for (int t = 0; t < 4; ++t) { float d = vals[t] - mean; vs += d * d; }
#pragma unroll
  for (int o = 16; o; o >>= 1) vs += __shfl_xor(vs, o);
  float rstd = rsqrtf(vs * (1.0f / 128.0f) + 1e-5f);
#pragma unroll
  for (int t = 0; t < 4; ++t) {
    int   j  = j0 + t;
    float ov = (vals[t] - mean) * rstd * g[j] + b[j];
    hout[base + j] = ov;
    h16[base + j]  = (half_t)ov;
  }
}

// ---------------------------------------------------------- fused Q/K/V GEMM
// One 64x128 A tile staged via TDM feeds all three projections: 384 WMMAs per
// block per staged tile. Each of 8 waves owns one 16-col tile.
__global__ __launch_bounds__(256) void qkv_kernel(
    const half_t* __restrict__ A, const half_t* __restrict__ Wqp,
    const half_t* __restrict__ Wkp, const half_t* __restrict__ Wvp,
    const float* __restrict__ bq, const float* __restrict__ bk,
    const float* __restrict__ bv, half_t* __restrict__ Oq,
    half_t* __restrict__ Ok, half_t* __restrict__ Ov) {
  extern __shared__ half_t Atile[];  // 64 * 128 halves = 16 KB
  int lane = threadIdx.x & 31, wave = threadIdx.x >> 5;
  int r0 = blockIdx.x * 64;
  stage_rows_to_lds(Atile, A + (size_t)r0 * D, 64);
  __syncthreads();
  int hl = lane >> 4, nn = lane & 15;
  const half_t* Wps[3] = {Wqp, Wkp, Wvp};
  const float*  bs[3]  = {bq, bk, bv};
  half_t*       Os[3]  = {Oq, Ok, Ov};
#pragma unroll
  for (int w = 0; w < 3; ++w) {
    v8f acc[4] = {};
#pragma unroll
    for (int kt = 0; kt < 4; ++kt) {
      v16h b = load_b_packed(Wps[w], kt * 8 + wave, lane);
#pragma unroll
      for (int rt = 0; rt < 4; ++rt) {
        v16h a = load_a_lds(Atile, D, rt * 16, kt * 32, lane);
        acc[rt] = wmma_f16(a, b, acc[rt]);
      }
    }
    int n = wave * 16 + nn;
    float bb = bs[w][n];
#pragma unroll
    for (int rt = 0; rt < 4; ++rt)
#pragma unroll
      for (int i = 0; i < 8; ++i) {
        int m = r0 + rt * 16 + i + 8 * hl;
        Os[w][(size_t)m * D + n] = (half_t)(acc[rt][i] + bb);
      }
  }
}

// ------------------------------------------------------------------ WMMA GEMM
// C[N,128] = A[N,128](f16) @ Wpacked + bias (used for the O projection).
template <bool HALF_OUT>
__global__ __launch_bounds__(256) void gemm128_kernel(
    const half_t* __restrict__ A, const half_t* __restrict__ Wp,
    const float* __restrict__ bias, float* __restrict__ Cf,
    half_t* __restrict__ Ch) {
  extern __shared__ half_t Atile[];  // 64 * 128 halves = 16 KB
  int lane = threadIdx.x & 31, wave = threadIdx.x >> 5;
  int r0 = blockIdx.x * 64;
  stage_rows_to_lds(Atile, A + (size_t)r0 * D, 64);
  __syncthreads();
  int hl = lane >> 4, nn = lane & 15;
  v8f acc[4] = {};
#pragma unroll
  for (int kt = 0; kt < 4; ++kt) {
    v16h b = load_b_packed(Wp, kt * 8 + wave, lane);
#pragma unroll
    for (int rt = 0; rt < 4; ++rt) {
      v16h a = load_a_lds(Atile, D, rt * 16, kt * 32, lane);
      acc[rt] = wmma_f16(a, b, acc[rt]);
    }
  }
  int n = wave * 16 + nn;
  float bb = bias[n];
#pragma unroll
  for (int rt = 0; rt < 4; ++rt)
#pragma unroll
    for (int i = 0; i < 8; ++i) {
      int   m = r0 + rt * 16 + i + 8 * hl;
      float v = acc[rt][i] + bb;
      if (HALF_OUT) Ch[(size_t)m * D + n] = (half_t)v;
      else          Cf[(size_t)m * D + n] = v;
    }
}

// --------------------------------------------------------------- fused FFN
// out[16,128] = relu(h16[16,128] @ W1 + b1) @ W2 + b2; A tile TDM-staged,
// intermediate T kept in LDS.
__global__ __launch_bounds__(256) void ffn_kernel(
    const half_t* __restrict__ h16, const half_t* __restrict__ W1p,
    const float* __restrict__ b1, const half_t* __restrict__ W2p,
    const float* __restrict__ b2, float* __restrict__ out) {
  extern __shared__ half_t smem[];
  half_t* Atile = smem;            // 16 * 128
  half_t* T     = smem + 16 * D;   // 16 * 512
  int lane = threadIdx.x & 31, wave = threadIdx.x >> 5;
  int r0 = blockIdx.x * 16;
  int hl = lane >> 4, nn = lane & 15;
  stage_rows_to_lds(Atile, h16 + (size_t)r0 * D, 16);
  __syncthreads();
  // phase 1: T = relu(A @ W1 + b1), each wave does 4 column tiles of 16
#pragma unroll
  for (int s = 0; s < 4; ++s) {
    int nt = wave * 4 + s;
    v8f c = {};
#pragma unroll
    for (int kt = 0; kt < 4; ++kt) {
      v16h a = load_a_lds(Atile, D, 0, kt * 32, lane);
      v16h b = load_b_packed(W1p, kt * 32 + nt, lane);
      c = wmma_f16(a, b, c);
    }
    int n = nt * 16 + nn;
    float bb = b1[n];
#pragma unroll
    for (int i = 0; i < 8; ++i) {
      float v = c[i] + bb;
      v = v > 0.0f ? v : 0.0f;
      T[(i + 8 * hl) * DFF + n] = (half_t)v;
    }
  }
  __syncthreads();
  // phase 2: out = T @ W2 + b2, each wave one 16-col tile, K=512 from LDS
  v8f c = {};
#pragma unroll
  for (int kt = 0; kt < 16; ++kt) {
    v16h a = load_a_lds(T, DFF, 0, kt * 32, lane);
    v16h b = load_b_packed(W2p, kt * 8 + wave, lane);
    c = wmma_f16(a, b, c);
  }
  int n = wave * 16 + nn;
  float bb = b2[n];
#pragma unroll
  for (int i = 0; i < 8; ++i)
    out[(size_t)(r0 + i + 8 * hl) * D + n] = c[i] + bb;
}

// ------------------------------------------------------------ edge attention
__device__ inline unsigned f2ord(float f) {
  unsigned u = __float_as_uint(f);
  return (u & 0x80000000u) ? ~u : (u | 0x80000000u);
}
__device__ inline float ord2f(unsigned u) {
  return (u & 0x80000000u) ? __uint_as_float(u ^ 0x80000000u)
                           : __uint_as_float(~u);
}

__global__ __launch_bounds__(256) void edge_score_kernel(
    const half_t* __restrict__ q16, const half_t* __restrict__ k16,
    const int* __restrict__ src, const int* __restrict__ dst,
    float* __restrict__ score, unsigned* __restrict__ mmax) {
  int t = blockIdx.x * 256 + threadIdx.x;
  if (t >= EDGES * H) return;
  int e = t >> 3, hh = t & 7;
  int s = src[e], d = dst[e];
  const uint4* qp = reinterpret_cast<const uint4*>(q16 + (size_t)d * D + hh * DK);
  const uint4* kp = reinterpret_cast<const uint4*>(k16 + (size_t)s * D + hh * DK);
  H16U qa, ka;
  qa.q[0] = qp[0]; qa.q[1] = qp[1];
  ka.q[0] = kp[0]; ka.q[1] = kp[1];
  float acc = 0.0f;
#pragma unroll
  for (int i = 0; i < DK; ++i) acc += (float)qa.h[i] * (float)ka.h[i];
  acc *= 0.25f;  // 1/sqrt(DK)
  score[t] = acc;
  atomicMax(mmax + (size_t)d * H + hh, f2ord(acc));
}

__global__ __launch_bounds__(256) void edge_accum_kernel(
    const half_t* __restrict__ v16buf, const int* __restrict__ src,
    const int* __restrict__ dst, const float* __restrict__ score,
    const unsigned* __restrict__ mmax, float* __restrict__ den,
    float* __restrict__ agg) {
  int t = blockIdx.x * 256 + threadIdx.x;
  if (t >= EDGES * H) return;
  int e = t >> 3, hh = t & 7;
  int s = src[e], d = dst[e];
  float mf = ord2f(mmax[(size_t)d * H + hh]);
  float ex = __expf(score[t] - mf);
  atomicAdd(den + (size_t)d * H + hh, ex);
  const uint4* vp =
      reinterpret_cast<const uint4*>(v16buf + (size_t)s * D + hh * DK);
  H16U va;
  va.q[0] = vp[0]; va.q[1] = vp[1];
  float* ap = agg + (size_t)d * D + hh * DK;
#pragma unroll
  for (int i = 0; i < DK; ++i) atomicAdd(ap + i, ex * (float)va.h[i]);
}

__global__ __launch_bounds__(256) void attn_final_kernel(
    const float* __restrict__ agg, const float* __restrict__ den,
    half_t* __restrict__ agg16) {
  int t = blockIdx.x * 256 + threadIdx.x;
  if (t >= NN * D) return;
  int n = t >> 7, j = t & 127, hh = j >> 4;
  agg16[t] = (half_t)(agg[t] / den[(size_t)n * H + hh]);
}

// ------------------------------------------------------------------- readout
__global__ __launch_bounds__(256) void readout_kernel(
    const float* __restrict__ h, const int* __restrict__ rid,
    const float* __restrict__ gw, const float* __restrict__ gb,
    float* __restrict__ out) {
  int lane = threadIdx.x & 31, wave = threadIdx.x >> 5;
  int r = blockIdx.x * 8 + wave;
  if (r >= NREAD) return;
  int node = rid[r];
  float acc[NCLS];
#pragma unroll
  for (int c = 0; c < NCLS; ++c) acc[c] = 0.0f;
  int j0 = lane * 4;
#pragma unroll
  for (int t = 0; t < 4; ++t) {
    int   j  = j0 + t;
    float hv = h[(size_t)node * D + j];
#pragma unroll
    for (int c = 0; c < NCLS; ++c) acc[c] += hv * gw[j * NCLS + c];
  }
#pragma unroll
  for (int c = 0; c < NCLS; ++c)
#pragma unroll
    for (int o = 16; o; o >>= 1) acc[c] += __shfl_xor(acc[c], o);
  if (lane == 0) {
    float mx = -1e30f;
#pragma unroll
    for (int c = 0; c < NCLS; ++c) { acc[c] += gb[c]; mx = fmaxf(mx, acc[c]); }
    float se = 0.0f;
#pragma unroll
    for (int c = 0; c < NCLS; ++c) se += __expf(acc[c] - mx);
    float lse = __logf(se);
#pragma unroll
    for (int c = 0; c < NCLS; ++c) out[(size_t)r * NCLS + c] = acc[c] - mx - lse;
  }
}

// -------------------------------------------------------------------- driver
extern "C" void kernel_launch(void* const* d_in, const int* in_sizes, int n_in,
                              void* d_out, int out_size, void* d_ws,
                              size_t ws_size, hipStream_t stream) {
  const int*   x        = (const int*)d_in[0];
  const int*   pos      = (const int*)d_in[1];
  const int*   leaf_ids = (const int*)d_in[2];
  const int*   src      = (const int*)d_in[3];
  const int*   dst      = (const int*)d_in[4];
  const int*   rid      = (const int*)d_in[5];
  // d_in[6] = h0 (zeros, re-created locally)
  const float* et   = (const float*)d_in[7];
  const float* embg = (const float*)d_in[8];
  const float* embb = (const float*)d_in[9];
  const float* Wq   = (const float*)d_in[10];
  const float* bq   = (const float*)d_in[11];
  const float* Wk   = (const float*)d_in[12];
  const float* bk   = (const float*)d_in[13];
  const float* Wv   = (const float*)d_in[14];
  const float* bv   = (const float*)d_in[15];
  const float* Wo   = (const float*)d_in[16];
  const float* bo   = (const float*)d_in[17];
  const float* ln1g = (const float*)d_in[18];
  const float* ln1b = (const float*)d_in[19];
  const float* W1   = (const float*)d_in[20];
  const float* b1   = (const float*)d_in[21];
  const float* W2   = (const float*)d_in[22];
  const float* b2   = (const float*)d_in[23];
  const float* ln2g = (const float*)d_in[24];
  const float* ln2b = (const float*)d_in[25];
  const float* genW = (const float*)d_in[26];
  const float* genb = (const float*)d_in[27];
  float* out = (float*)d_out;

  char*  base = (char*)d_ws;
  size_t off  = 0;
  auto take = [&](size_t bytes) -> void* {
    void* p = base + off;
    off = (off + bytes + 255) & ~(size_t)255;
    return p;
  };
  float*    h     = (float*)take((size_t)NN * D * 4);
  half_t*   h16   = (half_t*)take((size_t)NN * D * 2);
  half_t*   q16   = (half_t*)take((size_t)NN * D * 2);
  half_t*   k16   = (half_t*)take((size_t)NN * D * 2);
  half_t*   v16b  = (half_t*)take((size_t)NN * D * 2);
  float*    agg   = (float*)take((size_t)NN * D * 4);
  half_t*   agg16 = (half_t*)take((size_t)NN * D * 2);
  float*    tmpf  = (float*)take((size_t)NN * D * 4);
  unsigned* mmax  = (unsigned*)take((size_t)NN * H * 4);
  float*    den   = (float*)take((size_t)NN * H * 4);
  float*    score = (float*)take((size_t)EDGES * H * 4);
  half_t*   wqp   = (half_t*)take((size_t)NL * D * D * 2);
  half_t*   wkp   = (half_t*)take((size_t)NL * D * D * 2);
  half_t*   wvp   = (half_t*)take((size_t)NL * D * D * 2);
  half_t*   wop   = (half_t*)take((size_t)NL * D * D * 2);
  half_t*   w1p   = (half_t*)take((size_t)NL * D * DFF * 2);
  half_t*   w2p   = (half_t*)take((size_t)NL * DFF * D * 2);

  // pack weights to WMMA-B fragment layout (f16)
  for (int i = 0; i < NL; ++i) {
    pack_w_kernel<<<(D * D + 255) / 256, 256, 0, stream>>>(
        Wq + (size_t)i * D * D, wqp + (size_t)i * D * D, D, D);
    pack_w_kernel<<<(D * D + 255) / 256, 256, 0, stream>>>(
        Wk + (size_t)i * D * D, wkp + (size_t)i * D * D, D, D);
    pack_w_kernel<<<(D * D + 255) / 256, 256, 0, stream>>>(
        Wv + (size_t)i * D * D, wvp + (size_t)i * D * D, D, D);
    pack_w_kernel<<<(D * D + 255) / 256, 256, 0, stream>>>(
        Wo + (size_t)i * D * D, wop + (size_t)i * D * D, D, D);
    pack_w_kernel<<<(D * DFF + 255) / 256, 256, 0, stream>>>(
        W1 + (size_t)i * D * DFF, w1p + (size_t)i * D * DFF, D, DFF);
    pack_w_kernel<<<(DFF * D + 255) / 256, 256, 0, stream>>>(
        W2 + (size_t)i * DFF * D, w2p + (size_t)i * DFF * D, DFF, D);
  }

  // h = scatter(LN(embed + pe)) with internal nodes zeroed
  zero_pair_kernel<<<(NN * D) / 256, 256, 0, stream>>>(h, h16, NN * D);
  embed_kernel<<<NLEAF / 8, 256, 0, stream>>>(x, pos, leaf_ids, et, embg, embb,
                                              h, h16);

  const int    gemm_grid = NN / 64;
  const size_t gemm_lds  = (size_t)64 * D * 2;               // 16 KB A tile
  const size_t ffn_lds   = (size_t)(16 * D + 16 * DFF) * 2;  // 4 KB + 16 KB
  const int    edge_grid = (EDGES * H + 255) / 256;
  for (int i = 0; i < NL; ++i) {
    const size_t wo = (size_t)i * D * D;
    // fused Q,K,V projections: one staged A tile feeds 3 GEMMs
    qkv_kernel<<<gemm_grid, 256, gemm_lds, stream>>>(
        h16, wqp + wo, wkp + wo, wvp + wo, bq + (size_t)i * D,
        bk + (size_t)i * D, bv + (size_t)i * D, q16, k16, v16b);
    // segment-softmax attention over edges
    zero_u32_kernel<<<(NN * H) / 256, 256, 0, stream>>>(mmax, NN * H);
    zero_f32_kernel<<<(NN * H) / 256, 256, 0, stream>>>(den, NN * H);
    zero_f32_kernel<<<(NN * D) / 256, 256, 0, stream>>>(agg, NN * D);
    edge_score_kernel<<<edge_grid, 256, 0, stream>>>(q16, k16, src, dst, score,
                                                     mmax);
    edge_accum_kernel<<<edge_grid, 256, 0, stream>>>(v16b, src, dst, score,
                                                     mmax, den, agg);
    attn_final_kernel<<<(NN * D) / 256, 256, 0, stream>>>(agg, den, agg16);
    // output projection + residual LN
    gemm128_kernel<false><<<gemm_grid, 256, gemm_lds, stream>>>(
        agg16, wop + wo, bo + (size_t)i * D, tmpf, nullptr);
    resln_kernel<<<NN / 8, 256, 0, stream>>>(h, tmpf, ln1g + (size_t)i * D,
                                             ln1b + (size_t)i * D, h, h16);
    // fused FFN + residual LN
    ffn_kernel<<<NN / 16, 256, ffn_lds, stream>>>(
        h16, w1p + (size_t)i * D * DFF, b1 + (size_t)i * DFF,
        w2p + (size_t)i * DFF * D, b2 + (size_t)i * D, tmpf);
    resln_kernel<<<NN / 8, 256, 0, stream>>>(h, tmpf, ln2g + (size_t)i * D,
                                             ln2b + (size_t)i * D, h, h16);
  }

  readout_kernel<<<NREAD / 8, 256, 0, stream>>>(h, rid, genW, genb, out);
}
